// GraphAttn_42984032698880
// MI455X (gfx1250) — compile-verified, matrix operand order checked by reading, not run on hardware
//
#include <hip/hip_runtime.h>
#include <math.h>

#define N_NODES   50000
#define N_EDGES   800000
#define L_IN      1024
#define D_HID     256

typedef __attribute__((ext_vector_type(16))) __bf16 v16bf;
typedef __attribute__((ext_vector_type(4)))  __bf16 v4bf;
typedef __attribute__((ext_vector_type(8)))  float  v8f;

// ---------- helpers ----------
__device__ __forceinline__ unsigned fkey(float f) {
    unsigned u = __float_as_uint(f);
    return (u & 0x80000000u) ? ~u : (u | 0x80000000u);
}
__device__ __forceinline__ float funkey(unsigned k) {
    unsigned u = (k & 0x80000000u) ? (k & 0x7FFFFFFFu) : ~k;
    return __uint_as_float(u);
}
__device__ __forceinline__ void atomAddF(float* p, float v) {
    __hip_atomic_fetch_add(p, v, __ATOMIC_RELAXED, __HIP_MEMORY_SCOPE_AGENT);
}
__device__ __forceinline__ void atomMaxU(unsigned* p, unsigned v) {
    __hip_atomic_fetch_max(p, v, __ATOMIC_RELAXED, __HIP_MEMORY_SCOPE_AGENT);
}

// LDS K-contiguous tile stride: 40 halves = 80B -> 16B aligned chunks,
// 80B row stride is conflict-free across 16 lanes (5 coprime with 16).
#define ASTR 40

union Frag { v16bf v; uint4 u[2]; };

// ================= prep kernels =================
__global__ __launch_bounds__(256)
void cvt_bf16_kernel(const float* __restrict__ in, __bf16* __restrict__ out, int n4)
{
    int i = blockIdx.x * 256 + threadIdx.x;
    if (i >= n4) return;
    float4 f = ((const float4*)in)[i];
    v4bf o;
    o.x = (__bf16)f.x; o.y = (__bf16)f.y; o.z = (__bf16)f.z; o.w = (__bf16)f.w;
    ((v4bf*)out)[i] = o;
}

// Wt[n*K + k] = bf16(W[k*N + n])
__global__ __launch_bounds__(256)
void wt_kernel(const float* __restrict__ W, __bf16* __restrict__ Wt, int K, int N)
{
    int idx = blockIdx.x * 256 + threadIdx.x;
    if (idx >= K * N) return;
    int k = idx / N, n = idx - k * N;
    Wt[(size_t)n * K + k] = (__bf16)W[idx];
}

// ========== fused 4-way WMMA GEMM: {q,k,v,s} = x @ {Wq,Wk,Wv,Ws} + b ==========
// Xb: bf16 [M][1024]; Wt: bf16 [4][256][1024] (n-major, K-contiguous)
// Block tile 64x128, 4 waves, each wave 32x64 (2x4 WMMA tiles).
__global__ __launch_bounds__(128)
void gemm_x4_kernel(const __bf16* __restrict__ Xb, const __bf16* __restrict__ Wt,
                    const float* __restrict__ bq, const float* __restrict__ bk,
                    const float* __restrict__ bv, const float* __restrict__ bs,
                    float* __restrict__ oq, float* __restrict__ ok,
                    float* __restrict__ ov, float* __restrict__ os, int M)
{
    __shared__ __bf16 sA[64  * ASTR];   // 64 rows  x 32 k
    __shared__ __bf16 sB[128 * ASTR];   // 128 cols x 32 k
    const int K = L_IN;

    const int t    = threadIdx.x;
    const int lane = t & 31;
    const int wid  = t >> 5;
    const int wm   = (wid >> 1) * 32;   // 0,32
    const int wn   = (wid & 1) * 64;    // 0,64
    const int rowBase = blockIdx.x * 64;
    const int mat     = blockIdx.y >> 1;
    const int colBase = (blockIdx.y & 1) * 128;
    const __bf16* Wm  = Wt + (size_t)mat * D_HID * L_IN;

    const int lr = lane & 15;
    const int kb = (lane >> 4) * 8;     // K-half select (ISA 16-bit layout)

    v8f acc[2][4] = {};

    for (int k0 = 0; k0 < K; k0 += 32) {
        // stage A (64 rows x 32 k): branchless row clamp (OOB rows never stored)
        #pragma unroll
        for (int i = 0; i < 2; ++i) {
            int u  = t + i * 128;       // 0..255
            int r  = u >> 2;            // 0..63
            int c8 = (u & 3) * 8;       // 0,8,16,24
            int gr = min(rowBase + r, M - 1);
            uint4 d = *(const uint4*)(Xb + (size_t)gr * K + k0 + c8);
            *(uint4*)(&sA[r * ASTR + c8]) = d;
        }
        // stage B (128 cols x 32 k)
        #pragma unroll
        for (int i = 0; i < 4; ++i) {
            int u  = t + i * 128;       // 0..511
            int r  = u >> 2;            // 0..127
            int c8 = (u & 3) * 8;
            uint4 w = *(const uint4*)(Wm + (size_t)(colBase + r) * K + k0 + c8);
            *(uint4*)(&sB[r * ASTR + c8]) = w;
        }
        __syncthreads();

        Frag afr[2], bfr[4];
        #pragma unroll
        for (int s2 = 0; s2 < 2; ++s2) {
            const int arow = wm + s2 * 16 + lr;
            afr[s2].u[0] = *(const uint4*)(&sA[arow * ASTR + kb]);
            afr[s2].u[1] = *(const uint4*)(&sA[arow * ASTR + kb + 16]);
        }
        #pragma unroll
        for (int s2 = 0; s2 < 4; ++s2) {
            const int bcol = wn + s2 * 16 + lr;
            bfr[s2].u[0] = *(const uint4*)(&sB[bcol * ASTR + kb]);
            bfr[s2].u[1] = *(const uint4*)(&sB[bcol * ASTR + kb + 16]);
        }
        #pragma unroll
        for (int mi = 0; mi < 2; ++mi)
            #pragma unroll
            for (int ni = 0; ni < 4; ++ni)
                acc[mi][ni] = __builtin_amdgcn_wmma_f32_16x16x32_bf16(
                    false, afr[mi].v, false, bfr[ni].v, (short)0, acc[mi][ni], false, false);
        __syncthreads();
    }

    const float* bias = (mat == 0) ? bq : (mat == 1) ? bk : (mat == 2) ? bv : bs;
    float*       C    = (mat == 0) ? oq : (mat == 1) ? ok : (mat == 2) ? ov : os;

    const int cn = lane & 15;
    const int mo = (lane >> 4) * 8;
    #pragma unroll
    for (int mi = 0; mi < 2; ++mi) {
        #pragma unroll
        for (int ni = 0; ni < 4; ++ni) {
            const int col = colBase + wn + ni * 16 + cn;
            const float bv2 = bias[col];
            #pragma unroll
            for (int j = 0; j < 8; ++j) {
                const int row = rowBase + wm + mi * 16 + mo + j;
                if (row < M) C[(size_t)row * D_HID + col] = acc[mi][ni][j] + bv2;
            }
        }
    }
}

// ===== fused gate GEMM: g = tanh(h@Wa+ba) * sigmoid(h@Wb+bb), N=K=256 =====
// Hb: bf16 [M][256]; Wabt: bf16 [2][256][256] (n-major, K-contiguous)
__global__ __launch_bounds__(128)
void gemm_gate_kernel(const __bf16* __restrict__ Hb, const __bf16* __restrict__ Wabt,
                      const float* __restrict__ ba, const float* __restrict__ bb,
                      float* __restrict__ G, int M)
{
    __shared__ __bf16 sA [64 * ASTR];
    __shared__ __bf16 sBa[64 * ASTR];
    __shared__ __bf16 sBb[64 * ASTR];
    const int K = D_HID;

    const int t    = threadIdx.x;
    const int lane = t & 31;
    const int wid  = t >> 5;
    const int wm   = (wid >> 1) * 32;
    const int wn   = (wid & 1) * 32;
    const int rowBase = blockIdx.x * 64;
    const int colBase = blockIdx.y * 64;
    const __bf16* Wa = Wabt;
    const __bf16* Wb = Wabt + (size_t)D_HID * D_HID;

    const int lr = lane & 15;
    const int kb = (lane >> 4) * 8;

    v8f accA[2][2] = {};
    v8f accB[2][2] = {};

    for (int k0 = 0; k0 < K; k0 += 32) {
        #pragma unroll
        for (int i = 0; i < 2; ++i) {
            int u  = t + i * 128;
            int r  = u >> 2;
            int c8 = (u & 3) * 8;
            int gr = min(rowBase + r, M - 1);
            uint4 d = *(const uint4*)(Hb + (size_t)gr * K + k0 + c8);
            *(uint4*)(&sA[r * ASTR + c8]) = d;
            uint4 wa = *(const uint4*)(Wa + (size_t)(colBase + r) * K + k0 + c8);
            *(uint4*)(&sBa[r * ASTR + c8]) = wa;
            uint4 wb = *(const uint4*)(Wb + (size_t)(colBase + r) * K + k0 + c8);
            *(uint4*)(&sBb[r * ASTR + c8]) = wb;
        }
        __syncthreads();

        Frag afr[2], bfrA[2], bfrB[2];
        #pragma unroll
        for (int s2 = 0; s2 < 2; ++s2) {
            const int arow = wm + s2 * 16 + lr;
            afr[s2].u[0] = *(const uint4*)(&sA[arow * ASTR + kb]);
            afr[s2].u[1] = *(const uint4*)(&sA[arow * ASTR + kb + 16]);
            const int bcol = wn + s2 * 16 + lr;
            bfrA[s2].u[0] = *(const uint4*)(&sBa[bcol * ASTR + kb]);
            bfrA[s2].u[1] = *(const uint4*)(&sBa[bcol * ASTR + kb + 16]);
            bfrB[s2].u[0] = *(const uint4*)(&sBb[bcol * ASTR + kb]);
            bfrB[s2].u[1] = *(const uint4*)(&sBb[bcol * ASTR + kb + 16]);
        }
        #pragma unroll
        for (int mi = 0; mi < 2; ++mi)
            #pragma unroll
            for (int ni = 0; ni < 2; ++ni) {
                accA[mi][ni] = __builtin_amdgcn_wmma_f32_16x16x32_bf16(
                    false, afr[mi].v, false, bfrA[ni].v, (short)0, accA[mi][ni], false, false);
                accB[mi][ni] = __builtin_amdgcn_wmma_f32_16x16x32_bf16(
                    false, afr[mi].v, false, bfrB[ni].v, (short)0, accB[mi][ni], false, false);
            }
        __syncthreads();
    }

    const int cn = lane & 15;
    const int mo = (lane >> 4) * 8;
    #pragma unroll
    for (int mi = 0; mi < 2; ++mi) {
        #pragma unroll
        for (int ni = 0; ni < 2; ++ni) {
            const int col = colBase + wn + ni * 16 + cn;
            const float bva = ba[col];
            const float bvb = bb[col];
            #pragma unroll
            for (int j = 0; j < 8; ++j) {
                const int row = rowBase + wm + mi * 16 + mo + j;
                if (row < M) {
                    float va = accA[mi][ni][j] + bva;
                    float vb = accB[mi][ni][j] + bvb;
                    G[(size_t)row * D_HID + col] = tanhf(va) * (1.f / (1.f + __expf(-vb)));
                }
            }
        }
    }
}

// ================= edge phase =================
__global__ __launch_bounds__(256)
void edge_alpha_kernel(const float* __restrict__ q, const float* __restrict__ k,
                       const int* __restrict__ src, const int* __restrict__ dst,
                       float* __restrict__ alpha, unsigned* __restrict__ amaxKey, int E)
{
    int e = blockIdx.x * 8 + (threadIdx.x >> 5);
    if (e >= E) return;
    int lane = threadIdx.x & 31;
    int s = src[e], d = dst[e];
    const float4* qv = (const float4*)(q + (size_t)d * D_HID);
    const float4* kv = (const float4*)(k + (size_t)s * D_HID);
    float r = 0.f;
    #pragma unroll
    for (int i = 0; i < 2; ++i) {
        float4 a = qv[lane + i * 32];
        float4 b = kv[lane + i * 32];
        r += a.x * b.x + a.y * b.y + a.z * b.z + a.w * b.w;
    }
    #pragma unroll
    for (int off = 16; off > 0; off >>= 1) r += __shfl_xor(r, off, 32);
    if (lane == 0) {
        r *= 0.0625f;                     // 1/sqrt(256)
        alpha[e] = r;
        atomMaxU(&amaxKey[d], fkey(r));
    }
}

__global__ __launch_bounds__(256)
void edge_exp_kernel(const int* __restrict__ dst, const unsigned* __restrict__ amaxKey,
                     float* __restrict__ alpha_ex, float* __restrict__ den, int E)
{
    int e = blockIdx.x * 256 + threadIdx.x;
    if (e >= E) return;
    int d = dst[e];
    float ex = __expf(alpha_ex[e] - funkey(amaxKey[d]));
    alpha_ex[e] = ex;
    atomAddF(&den[d], ex);
}

__global__ __launch_bounds__(256)
void edge_aggregate_kernel(const float* __restrict__ v,
                           const int* __restrict__ src, const int* __restrict__ dst,
                           const float* __restrict__ ex, const float* __restrict__ den,
                           float* __restrict__ agg, int E)
{
    int e = blockIdx.x * 8 + (threadIdx.x >> 5);
    if (e >= E) return;
    int lane = threadIdx.x & 31;
    int s = src[e], d = dst[e];
    const float* vp = v + (size_t)s * D_HID;
    __builtin_prefetch(vp + lane * 8, 0, 1);   // global_prefetch_b8: hide gather latency
    float w = ex[e] / (den[d] + 1e-16f);
    float* ap = agg + (size_t)d * D_HID;
    #pragma unroll
    for (int i = 0; i < 8; ++i) {
        int c = lane + i * 32;
        atomAddF(&ap[c], w * vp[c]);
    }
}

// ================= h = gelu(agg + s), also emit bf16 copy =================
__global__ __launch_bounds__(256)
void gelu_kernel(const float* __restrict__ agg, const float* __restrict__ s,
                 float* __restrict__ h, __bf16* __restrict__ hb, int n)
{
    int i = blockIdx.x * 256 + threadIdx.x;
    if (i < n) {
        float o = agg[i] + s[i];
        float r = 0.5f * o * (1.f + erff(o * 0.70710678118654752f));
        h[i]  = r;
        hb[i] = (__bf16)r;
    }
}

// ================= scores + global softmax + readout =================
__global__ __launch_bounds__(256)
void scores_kernel(const float* __restrict__ g, const float* __restrict__ Wc,
                   const float* __restrict__ bc,
                   float* __restrict__ sc, unsigned* __restrict__ cmaxKey, int N)
{
    int n = blockIdx.x * 8 + (threadIdx.x >> 5);
    if (n >= N) return;
    int lane = threadIdx.x & 31;
    const float* gp = g + (size_t)n * D_HID;
    float r0 = 0.f, r1 = 0.f;
    #pragma unroll
    for (int i = 0; i < 8; ++i) {
        int c = lane + i * 32;
        float gv = gp[c];
        r0 += gv * Wc[c * 2 + 0];
        r1 += gv * Wc[c * 2 + 1];
    }
    #pragma unroll
    for (int off = 16; off > 0; off >>= 1) {
        r0 += __shfl_xor(r0, off, 32);
        r1 += __shfl_xor(r1, off, 32);
    }
    if (lane == 0) {
        r0 += bc[0]; r1 += bc[1];
        sc[n * 2 + 0] = r0;
        sc[n * 2 + 1] = r1;
        atomMaxU(&cmaxKey[0], fkey(r0));
        atomMaxU(&cmaxKey[1], fkey(r1));
    }
}

__global__ __launch_bounds__(256)
void softmax_exp_kernel(float* __restrict__ sc, const unsigned* __restrict__ cmaxKey,
                        float* __restrict__ csum, int N)
{
    __shared__ float ls[2];
    if (threadIdx.x < 2) ls[threadIdx.x] = 0.f;
    __syncthreads();
    int n = blockIdx.x * 256 + threadIdx.x;
    if (n < N) {
        float e0 = __expf(sc[n * 2 + 0] - funkey(cmaxKey[0]));
        float e1 = __expf(sc[n * 2 + 1] - funkey(cmaxKey[1]));
        sc[n * 2 + 0] = e0;
        sc[n * 2 + 1] = e1;
        atomicAdd(&ls[0], e0);
        atomicAdd(&ls[1], e1);
    }
    __syncthreads();
    if (threadIdx.x < 2) atomAddF(&csum[threadIdx.x], ls[threadIdx.x]);
}

__global__ __launch_bounds__(256)
void finalize_kernel(const float* __restrict__ e, const float* __restrict__ csum,
                     const float* __restrict__ h, const int* __restrict__ label,
                     float* __restrict__ y, float* __restrict__ A, int N)
{
    __shared__ float la0[256], la1[256];
    const int t = threadIdx.x;
    const float inv0 = 1.f / csum[0];
    const float inv1 = 1.f / csum[1];
    const int lab = label[0];
    float y0 = 0.f, y1 = 0.f;

    for (int base = blockIdx.x * 256; base < N; base += gridDim.x * 256) {
        int n = base + t;
        float a0 = 0.f, a1 = 0.f;
        if (n < N) {
            a0 = e[n * 2 + 0] * inv0;
            a1 = e[n * 2 + 1] * inv1;
            A[n] = (lab == 0) ? a0 : a1;
        }
        la0[t] = a0; la1[t] = a1;
        __syncthreads();
        int lim = min(256, N - base);
        for (int j = 0; j < lim; ++j) {
            float hv = h[(size_t)(base + j) * D_HID + t];
            y0 += la0[j] * hv;
            y1 += la1[j] * hv;
        }
        __syncthreads();
    }
    atomAddF(&y[t], y0);
    atomAddF(&y[D_HID + t], y1);
}

// ================= launch =================
extern "C" void kernel_launch(void* const* d_in, const int* in_sizes, int n_in,
                              void* d_out, int out_size, void* d_ws, size_t ws_size,
                              hipStream_t stream)
{
    const float* x  = (const float*)d_in[0];
    const float* Wq = (const float*)d_in[1];  const float* bq = (const float*)d_in[2];
    const float* Wk = (const float*)d_in[3];  const float* bk = (const float*)d_in[4];
    const float* Wv = (const float*)d_in[5];  const float* bv = (const float*)d_in[6];
    const float* Ws = (const float*)d_in[7];  const float* bs = (const float*)d_in[8];
    const float* Wa = (const float*)d_in[9];  const float* ba = (const float*)d_in[10];
    const float* Wb = (const float*)d_in[11]; const float* bb = (const float*)d_in[12];
    const float* Wc = (const float*)d_in[13]; const float* bc = (const float*)d_in[14];
    const int* edge_index = (const int*)d_in[15];
    const int* label      = (const int*)d_in[16];
    const int* src = edge_index;
    const int* dst = edge_index + N_EDGES;

    // outputs: y (2x256) | A (50000) | h (50000x256)
    float* out = (float*)d_out;
    float* y = out;
    float* A = out + 512;
    float* h = out + 512 + N_NODES;

    // workspace layout (floats)
    const size_t NND = (size_t)N_NODES * D_HID;   // 12.8M
    float* q   = (float*)d_ws;
    float* k   = q   + NND;
    float* v   = k   + NND;
    float* s   = v   + NND;
    float* agg = s   + NND;
    float*    alpha   = agg + NND;                     // N_EDGES
    float*    den     = alpha + N_EDGES;               // N_NODES
    unsigned* amaxKey = (unsigned*)(den + N_NODES);    // N_NODES
    float*    sc      = (float*)(amaxKey + N_NODES);   // 2*N_NODES
    unsigned* cmaxKey = (unsigned*)(sc + 2 * N_NODES); // 2
    float*    csum    = (float*)(cmaxKey + 2);         // 2
    __bf16*   xb      = (__bf16*)(csum + 2);           // 50000*1024 halves
    __bf16*   hb      = xb + (size_t)N_NODES * L_IN;   // 50000*256 halves
    __bf16*   Wt4     = hb + NND;                      // 4*256*1024 halves
    __bf16*   Wab     = Wt4 + (size_t)4 * D_HID * L_IN;// 2*256*256 halves
    float*    g       = q;  // q dead after edge_alpha; reuse for gate output

    // zero accumulators (uint-key 0 == "below any finite float")
    hipMemsetAsync(agg, 0, NND * sizeof(float), stream);
    hipMemsetAsync(den, 0, N_NODES * sizeof(float), stream);
    hipMemsetAsync(amaxKey, 0, N_NODES * sizeof(unsigned), stream);
    hipMemsetAsync(cmaxKey, 0, 2 * sizeof(unsigned), stream);
    hipMemsetAsync(csum, 0, 2 * sizeof(float), stream);
    hipMemsetAsync(y, 0, 512 * sizeof(float), stream);

    // prep: bf16 conversions / weight transposes
    const int xN4 = (int)((size_t)N_NODES * L_IN / 4);             // 12.8M float4s
    cvt_bf16_kernel<<<(xN4 + 255) / 256, 256, 0, stream>>>(x, xb, xN4);
    const int wElems = L_IN * D_HID;                               // 262144
    wt_kernel<<<(wElems + 255) / 256, 256, 0, stream>>>(Wq, Wt4 + 0 * (size_t)wElems, L_IN, D_HID);
    wt_kernel<<<(wElems + 255) / 256, 256, 0, stream>>>(Wk, Wt4 + 1 * (size_t)wElems, L_IN, D_HID);
    wt_kernel<<<(wElems + 255) / 256, 256, 0, stream>>>(Wv, Wt4 + 2 * (size_t)wElems, L_IN, D_HID);
    wt_kernel<<<(wElems + 255) / 256, 256, 0, stream>>>(Ws, Wt4 + 3 * (size_t)wElems, L_IN, D_HID);
    const int gElems = D_HID * D_HID;                              // 65536
    wt_kernel<<<(gElems + 255) / 256, 256, 0, stream>>>(Wa, Wab + 0 * (size_t)gElems, D_HID, D_HID);
    wt_kernel<<<(gElems + 255) / 256, 256, 0, stream>>>(Wb, Wab + 1 * (size_t)gElems, D_HID, D_HID);

    // fused q/k/v/s GEMM: x read once; grid.y: mat(4) x colBlock(2 of 128)
    gemm_x4_kernel<<<dim3((N_NODES + 63) / 64, 8), 128, 0, stream>>>(
        xb, Wt4, bq, bk, bv, bs, q, k, v, s, N_NODES);

    edge_alpha_kernel<<<N_EDGES / 8, 256, 0, stream>>>(q, k, src, dst, alpha, amaxKey, N_EDGES);
    edge_exp_kernel<<<(N_EDGES + 255) / 256, 256, 0, stream>>>(dst, amaxKey, alpha, den, N_EDGES);
    edge_aggregate_kernel<<<N_EDGES / 8, 256, 0, stream>>>(v, src, dst, alpha, den, agg, N_EDGES);

    gelu_kernel<<<(int)((NND + 255) / 256), 256, 0, stream>>>(agg, s, h, hb, (int)NND);

    gemm_gate_kernel<<<dim3((N_NODES + 63) / 64, 4), 128, 0, stream>>>(
        hb, Wab, ba, bb, g, N_NODES);

    scores_kernel<<<(N_NODES + 7) / 8, 256, 0, stream>>>(g, Wc, bc, sc, cmaxKey, N_NODES);
    softmax_exp_kernel<<<(N_NODES + 255) / 256, 256, 0, stream>>>(sc, cmaxKey, csum, N_NODES);
    finalize_kernel<<<200, 256, 0, stream>>>(sc, csum, h, label, y, A, N_NODES);
}
